// MultiHeadSelfAttention_85366769975734
// MI455X (gfx1250) — matrix operand order repaired and checked by reading, not compiled, not run on hardware
//
#include <hip/hip_runtime.h>
#include <math.h>

// ---------------------------------------------------------------------------
// Multi-head self attention for MI455X (gfx1250, wave32, WMMA).
// f32 -> bf16, GEMMs via v_wmma_f32_16x16x32_bf16 (f32 accumulate), async
// global->LDS double-buffered weight tiles, register-resident flash attention.
// ---------------------------------------------------------------------------

typedef __attribute__((ext_vector_type(16))) __bf16 v16bf;
typedef __attribute__((ext_vector_type(8)))  __bf16 v8bf;
typedef __attribute__((ext_vector_type(8)))  float  v8f;
typedef __attribute__((ext_vector_type(4)))  int    v4i;

#define WMMA_BF16(A_, B_, C_) \
  __builtin_amdgcn_wmma_f32_16x16x32_bf16(false, (A_), false, (B_), (short)0, (C_), false, false)

#define AS1 __attribute__((address_space(1)))
#define AS3 __attribute__((address_space(3)))

#if defined(__gfx1250__) && __has_builtin(__builtin_amdgcn_global_load_async_to_lds_b128) && \
    __has_builtin(__builtin_amdgcn_s_wait_asynccnt)
#define USE_ASYNC_LDS 1
#endif

// Problem constants (fixed by reference)
#define BB    2
#define SS_   2048
#define DD_   1024
#define HH_   16
#define HD__  64
#define BSn   (BB * SS_)          // 4096 rows

// ---------------------------------------------------------------------------
// Fragment loaders (contiguous k order, same for A and B)
// A fragment 16x32 bf16: lane<16: M=lane, K = k0+0..15
//                        lane>=16: M=lane-16, K = k0+16..31
// ---------------------------------------------------------------------------
__device__ __forceinline__ v16bf ld_afrag(const __bf16* __restrict__ src,
                                          int row0, int ld, int k0, int lane) {
  const int m = lane & 15, half = lane >> 4;
  const __bf16* p = src + (size_t)(row0 + m) * ld;
  v8bf lo = *(const v8bf*)(p + k0 + half * 16);
  v8bf hi = *(const v8bf*)(p + k0 + half * 16 + 8);
  return __builtin_shufflevector(lo, hi, 0, 1, 2, 3, 4, 5, 6, 7,
                                 8, 9, 10, 11, 12, 13, 14, 15);
}

// B fragment 32x16 bf16 from a row-major [N,K] matrix (computing A @ B^T):
// lane<16: N=n0+lane, K=k0+0..15 ; lane>=16: N=n0+lane-16, K=k0+16..31
__device__ __forceinline__ v16bf ld_bfrag(const __bf16* __restrict__ src,
                                          int n0, int ld, int k0, int lane) {
  const __bf16* p = src + (size_t)(n0 + (lane & 15)) * ld + k0 + (lane >> 4) * 16;
  return *(const v16bf*)p;
}

// ---------------------------------------------------------------------------
// f32 -> bf16 conversion
// ---------------------------------------------------------------------------
__global__ __launch_bounds__(256) void cvt_bf16_kernel(const float* __restrict__ in,
                                                       __bf16* __restrict__ out, int n) {
  int i = blockIdx.x * 256 + threadIdx.x;
  if (i < n) out[i] = (__bf16)in[i];
}

// ---------------------------------------------------------------------------
// GEMM: C[M,N] = A[M,K] @ W[N,K]^T + bias[N]
// Block = 256 threads (8 waves). Block tile 256(M) x 64(N), wave tile 32 x 64
// (8 WMMAs per k-step). W tile [64n x 32k] double-buffered in LDS via async
// global->LDS copies (ASYNCcnt) when available.
// mode 0: QKV epilogue -> scatter Q[B,H,S,64], K[B,H,S,64], V^T[B,H,64,S] (bf16)
// mode 1: fp32 epilogue -> fo[M,N]
// ---------------------------------------------------------------------------
__global__ __launch_bounds__(256) void gemm_bf16_kernel(
    const __bf16* __restrict__ A, const __bf16* __restrict__ W,
    const float* __restrict__ bias,
    __bf16* __restrict__ qo, __bf16* __restrict__ ko, __bf16* __restrict__ vto,
    float* __restrict__ fo, int M, int N, int K, int mode) {
  __shared__ __attribute__((aligned(64))) __bf16 bs[2][64 * 32];  // 2 x 4KB

  const int tid  = threadIdx.x;
  const int lane = tid & 31;
  const int wv   = tid >> 5;
  const int m0   = blockIdx.y * 256 + wv * 32;
  const int n0   = blockIdx.x * 64;

  // Cooperative W-tile copy mapping: 256 threads x 16B = 64 rows x 64B
  const int row = tid >> 2, seg = tid & 3;
  const __bf16* wsrc = W + (size_t)(n0 + row) * K + seg * 8;
  const int lofs = row * 32 + seg * 8;

  // Prologue: stage tile k0=0 into buffer 0
#ifdef USE_ASYNC_LDS
  __builtin_amdgcn_global_load_async_to_lds_b128(
      (AS1 v4i*)wsrc, (AS3 v4i*)(&bs[0][lofs]), 0, 0);
  __builtin_amdgcn_s_wait_asynccnt(0);
#else
  *(v8bf*)(&bs[0][lofs]) = *(const v8bf*)wsrc;
#endif
  __syncthreads();

  v8f acc[2][4] = {};

  for (int k0 = 0; k0 < K; k0 += 32) {
    const int cur = (k0 >> 5) & 1;
    // Kick off next tile's copy into the other buffer before computing.
    if (k0 + 32 < K) {
#ifdef USE_ASYNC_LDS
      __builtin_amdgcn_global_load_async_to_lds_b128(
          (AS1 v4i*)(wsrc + k0 + 32), (AS3 v4i*)(&bs[cur ^ 1][lofs]), 0, 0);
#else
      *(v8bf*)(&bs[cur ^ 1][lofs]) = *(const v8bf*)(wsrc + k0 + 32);
#endif
    }

    v16bf af0 = ld_afrag(A, m0, K, k0, lane);
    v16bf af1 = ld_afrag(A, m0 + 16, K, k0, lane);
#pragma unroll
    for (int t = 0; t < 4; ++t) {
      const __bf16* p = &bs[cur][(t * 16 + (lane & 15)) * 32 + (lane >> 4) * 16];
      v16bf bfr = *(const v16bf*)p;
      acc[0][t] = WMMA_BF16(af0, bfr, acc[0][t]);
      acc[1][t] = WMMA_BF16(af1, bfr, acc[1][t]);
    }

#ifdef USE_ASYNC_LDS
    __builtin_amdgcn_s_wait_asynccnt(0);
#endif
    __syncthreads();
  }

#pragma unroll
  for (int a = 0; a < 2; ++a) {
    const int mbase = m0 + a * 16 + ((lane < 16) ? 0 : 8);
#pragma unroll
    for (int t = 0; t < 4; ++t) {
      const int n = n0 + t * 16 + (lane & 15);
      const float bv = bias[n];
#pragma unroll
      for (int r = 0; r < 8; ++r) {
        const float val = acc[a][t][r] + bv;
        const int m = mbase + r;
        if (mode == 1) {
          fo[(size_t)m * N + n] = val;
        } else {
          const int type = n >> 10;            // 0=Q 1=K 2=V
          const int jd   = n & 1023;
          const int h    = jd >> 6;
          const int d    = jd & 63;
          const int b    = m >> 11;            // / S
          const int s    = m & 2047;
          const size_t bh = (size_t)b * HH_ + h;
          const __bf16 bvx = (__bf16)val;
          if (type == 0)      qo[(bh * SS_ + s) * HD__ + d] = bvx;
          else if (type == 1) ko[(bh * SS_ + s) * HD__ + d] = bvx;
          else                vto[(bh * HD__ + d) * SS_ + s] = bvx;  // V transposed
        }
      }
    }
  }
}

// ---------------------------------------------------------------------------
// Flash attention, one wave per 16 query rows, 32 keys per iteration.
// Scores computed transposed: S^T = K * Q^T  (M=keys, N=queries) so the
// softmax key-reduction is VGPR-axis + one shfl_xor(16), and P^T feeds
// attended^T = V^T * P^T directly. No LDS, no barriers.
// ---------------------------------------------------------------------------
__global__ __launch_bounds__(256) void attn_kernel(
    const __bf16* __restrict__ Q, const __bf16* __restrict__ Km,
    const __bf16* __restrict__ Vt, __bf16* __restrict__ AO) {
  const int gw   = (blockIdx.x * 256 + threadIdx.x) >> 5;
  const int lane = threadIdx.x & 31;
  const int wph  = SS_ / 16;                 // 128 query tiles per (b,h)
  const int bh   = gw / wph;
  const int q0   = (gw % wph) * 16;

  const __bf16* Qp = Q  + (size_t)bh * SS_ * HD__;   // [S,64]
  const __bf16* Kp = Km + (size_t)bh * SS_ * HD__;   // [S,64]
  const __bf16* Vp = Vt + (size_t)bh * HD__ * SS_;   // [64,S] (transposed)

  // Q^T as two B fragments (d = 0..31, 32..63), kept in registers all loop.
  const v16bf qf0 = ld_bfrag(Qp, q0, HD__, 0, lane);
  const v16bf qf1 = ld_bfrag(Qp, q0, HD__, 32, lane);

  v8f acc[4] = {};                 // attended^T: 64(d) x 16(q)
  float mrun = -INFINITY, lrun = 0.f;
  const float scale = 0.125f;      // 1/sqrt(64)
  const bool lowh = (lane < 16);

  for (int kt = 0; kt < SS_; kt += 32) {
    // K rows as A fragments: two key halves x two d halves
    v16bf ka00 = ld_afrag(Kp, kt,      HD__, 0,  lane);
    v16bf ka01 = ld_afrag(Kp, kt,      HD__, 32, lane);
    v16bf ka10 = ld_afrag(Kp, kt + 16, HD__, 0,  lane);
    v16bf ka11 = ld_afrag(Kp, kt + 16, HD__, 32, lane);

    v8f s0 = {}, s1 = {};
    s0 = WMMA_BF16(ka00, qf0, s0);
    s0 = WMMA_BF16(ka01, qf1, s0);   // keys kt..kt+15   (rows), q (cols)
    s1 = WMMA_BF16(ka10, qf0, s1);
    s1 = WMMA_BF16(ka11, qf1, s1);   // keys kt+16..kt+31

    // Online softmax over the key axis (VGPR axis + lane-half exchange).
    float p0[8], p1[8];
    float mloc = -INFINITY;
#pragma unroll
    for (int r = 0; r < 8; ++r) {
      p0[r] = s0[r] * scale;
      p1[r] = s1[r] * scale;
      mloc = fmaxf(mloc, fmaxf(p0[r], p1[r]));
    }
    mloc = fmaxf(mloc, __shfl_xor(mloc, 16));
    const float mnew  = fmaxf(mrun, mloc);
    const float alpha = __expf(mrun - mnew);
    float lloc = 0.f;
#pragma unroll
    for (int r = 0; r < 8; ++r) {
      p0[r] = __expf(p0[r] - mnew);
      p1[r] = __expf(p1[r] - mnew);
      lloc += p0[r] + p1[r];
    }
    lloc += __shfl_xor(lloc, 16);
    lrun = lrun * alpha + lloc;
    mrun = mnew;
#pragma unroll
    for (int t = 0; t < 4; ++t)
#pragma unroll
      for (int r = 0; r < 8; ++r) acc[t][r] *= alpha;

    // Build P^T B-fragment (keys = K-dim 0..31, cols = q) via half exchange.
    v16bf pb;
#pragma unroll
    for (int r = 0; r < 8; ++r) {
      const float o0 = __shfl_xor(p0[r], 16);
      const float o1 = __shfl_xor(p1[r], 16);
      pb[r]     = (__bf16)(lowh ? p0[r] : o1);
      pb[r + 8] = (__bf16)(lowh ? o0 : p1[r]);
    }

    // attended^T += V^T (d x keys) * P^T (keys x q)
#pragma unroll
    for (int t = 0; t < 4; ++t) {
      v16bf vf = ld_afrag(Vp, t * 16, SS_, kt, lane);
      acc[t] = WMMA_BF16(vf, pb, acc[t]);
    }
  }

  const float inv = 1.f / lrun;
  const int b = bh / HH_, h = bh % HH_;
  const int q = q0 + (lane & 15);
  const int roff = lowh ? 0 : 8;
  const size_t rowbase = ((size_t)(b * SS_ + q)) * DD_ + h * HD__;
#pragma unroll
  for (int t = 0; t < 4; ++t)
#pragma unroll
    for (int r = 0; r < 8; ++r)
      AO[rowbase + t * 16 + roff + r] = (__bf16)(acc[t][r] * inv);
}

// ---------------------------------------------------------------------------
// Launch
// ---------------------------------------------------------------------------
extern "C" void kernel_launch(void* const* d_in, const int* in_sizes, int n_in,
                              void* d_out, int out_size, void* d_ws, size_t ws_size,
                              hipStream_t stream) {
  (void)in_sizes; (void)n_in; (void)out_size; (void)ws_size;
  const float* x     = (const float*)d_in[0];
  const float* w_qkv = (const float*)d_in[1];
  const float* b_qkv = (const float*)d_in[2];
  const float* w_out = (const float*)d_in[3];
  const float* b_out = (const float*)d_in[4];
  float* out = (float*)d_out;

  size_t off = 0;
  auto take = [&](size_t bytes) -> void* {
    void* p = (char*)d_ws + off;
    off += (bytes + 255) & ~(size_t)255;
    return p;
  };
  __bf16* xb  = (__bf16*)take((size_t)BSn * DD_ * 2);        // 8 MB
  __bf16* wqb = (__bf16*)take((size_t)3 * DD_ * DD_ * 2);    // 6 MB
  __bf16* wob = (__bf16*)take((size_t)DD_ * DD_ * 2);        // 2 MB
  __bf16* Qb  = (__bf16*)take((size_t)BSn * DD_ * 2);        // 8 MB
  __bf16* Kb  = (__bf16*)take((size_t)BSn * DD_ * 2);        // 8 MB
  __bf16* Vtb = (__bf16*)take((size_t)BSn * DD_ * 2);        // 8 MB
  __bf16* AOb = (__bf16*)take((size_t)BSn * DD_ * 2);        // 8 MB

  // 1) fp32 -> bf16 conversions
  {
    int n1 = BSn * DD_;
    cvt_bf16_kernel<<<(n1 + 255) / 256, 256, 0, stream>>>(x, xb, n1);
    int n2 = 3 * DD_ * DD_;
    cvt_bf16_kernel<<<(n2 + 255) / 256, 256, 0, stream>>>(w_qkv, wqb, n2);
    int n3 = DD_ * DD_;
    cvt_bf16_kernel<<<(n3 + 255) / 256, 256, 0, stream>>>(w_out, wob, n3);
  }

  // 2) QKV projection: [4096,1024] @ [3072,1024]^T + b_qkv -> Q,K,V^T (bf16)
  gemm_bf16_kernel<<<dim3((3 * DD_) / 64, BSn / 256), 256, 0, stream>>>(
      xb, wqb, b_qkv, Qb, Kb, Vtb, nullptr, BSn, 3 * DD_, DD_, 0);

  // 3) Flash attention: B*H*(S/16) waves / 8 waves per block = 512 blocks
  attn_kernel<<<(BB * HH_ * (SS_ / 16)) / 8, 256, 0, stream>>>(Qb, Kb, Vtb, AOb);

  // 4) Output projection: [4096,1024] @ [1024,1024]^T + b_out -> f32 d_out
  gemm_bf16_kernel<<<dim3(DD_ / 64, BSn / 256), 256, 0, stream>>>(
      AOb, wob, b_out, nullptr, nullptr, nullptr, out, BSn, DD_, DD_, 1);
}
